// OIMLoss_71622874628508
// MI455X (gfx1250) — matrix operand order repaired and checked by reading, not compiled, not run on hardware
//
#include <hip/hip_runtime.h>

// ---------------------------------------------------------------------------
// OIM loss, fused on gfx1250:
//   M=16384 pixels, N=5532 classes (pad 5568), K=256 features
//   loss = mean_p [ logsumexp_n(A_p . L_n) - A_p . L_target(p) ]
// Pipeline: (1) transpose+cvt inputs -> A_bf16[M][K]
//           (2) cvt+pad lut -> L_bf16[NPAD][K]
//           (3) per-pixel target logit (wave32 dot)
//           (4) fused WMMA GEMM + online softmax -> nll[M]
//               (double-buffered LDS fed by ASYNC global->LDS DMA, prefetch)
//           (5) deterministic single-block tree reduction -> d_out[0]
// ---------------------------------------------------------------------------

#define NUM_CLASSES 5532
#define NUM_FEAT    256
#define NPAD        5568        // 87 * 64
#define NTILES      (NPAD / 64) // 87
#define M_TOTAL     16384       // 4*64*64
#define HW          4096        // 64*64

typedef __attribute__((ext_vector_type(16))) __bf16 v16bf;
typedef __attribute__((ext_vector_type(8)))  __bf16 v8bf;
typedef __attribute__((ext_vector_type(8)))  float  v8f;

#if __has_builtin(__builtin_amdgcn_global_load_async_to_lds_b128)
#define OIM_ASYNC 1
#else
#define OIM_ASYNC 0
#endif

// Builtin signature (from hipcc diagnostic): params are int4* with AS(1)/AS(3).
typedef int v4i_gcc __attribute__((vector_size(4 * sizeof(int))));
typedef __attribute__((address_space(1))) v4i_gcc* gv4i_p;
typedef __attribute__((address_space(3))) v4i_gcc* lv4i_p;

// Copy one 32KB LUT tile (64 classes x 256 K, bf16) global -> LDS.
// 256 threads x 8 chunks x 16B. Async path: no VGPR staging, ASYNCcnt-tracked.
__device__ __forceinline__ void tile_copy(const __bf16* __restrict__ gsrc,
                                          __bf16* ldst, int tid) {
#if OIM_ASYNC
    gv4i_p g = (gv4i_p)((const char*)gsrc + tid * 16);
    lv4i_p l = (lv4i_p)((char*)ldst + tid * 16);
    __builtin_amdgcn_global_load_async_to_lds_b128(g, l, 0,     0);
    __builtin_amdgcn_global_load_async_to_lds_b128(g, l, 4096,  0);
    __builtin_amdgcn_global_load_async_to_lds_b128(g, l, 8192,  0);
    __builtin_amdgcn_global_load_async_to_lds_b128(g, l, 12288, 0);
    __builtin_amdgcn_global_load_async_to_lds_b128(g, l, 16384, 0);
    __builtin_amdgcn_global_load_async_to_lds_b128(g, l, 20480, 0);
    __builtin_amdgcn_global_load_async_to_lds_b128(g, l, 24576, 0);
    __builtin_amdgcn_global_load_async_to_lds_b128(g, l, 28672, 0);
#else
    const uint4* src = (const uint4*)gsrc;
    uint4* dst = (uint4*)ldst;
#pragma unroll
    for (int i = 0; i < 8; ++i) dst[tid + i * 256] = src[tid + i * 256];
#endif
}

__device__ __forceinline__ void tile_copy_wait() {
#if OIM_ASYNC
#if __has_builtin(__builtin_amdgcn_s_wait_asynccnt)
    __builtin_amdgcn_s_wait_asynccnt(0);
#else
    asm volatile("s_wait_asynccnt 0" ::: "memory");
#endif
#endif
}

// (1) inputs [4][256][4096] f32  ->  A [16384][256] bf16   (p = b*4096 + hw)
__global__ void oim_prep_a(const float* __restrict__ in, __bf16* __restrict__ A) {
    int idx = blockIdx.x * blockDim.x + threadIdx.x;      // over 4*256*4096
    if (idx >= 4 * NUM_FEAT * HW) return;
    int hw = idx & (HW - 1);
    int c  = (idx >> 12) & (NUM_FEAT - 1);
    int b  = idx >> 20;
    A[(size_t)(b * HW + hw) * NUM_FEAT + c] = (__bf16)in[idx];
}

// (2) lut [5532][256] f32 -> L [5568][256] bf16, zero-padded rows
__global__ void oim_prep_lut(const float* __restrict__ lut, __bf16* __restrict__ L) {
    int idx = blockIdx.x * blockDim.x + threadIdx.x;      // over NPAD*256
    if (idx >= NPAD * NUM_FEAT) return;
    int n = idx >> 8;
    float v = (n < NUM_CLASSES) ? lut[idx] : 0.0f;
    L[idx] = (__bf16)v;
}

// (3) t_logit[p] = dot(A[p], L[target[p]]) — one wave32 per pixel, shuffle tree
__global__ void oim_tlogit(const __bf16* __restrict__ A, const __bf16* __restrict__ L,
                           const int* __restrict__ tgt, float* __restrict__ tl) {
    int pix  = (blockIdx.x * blockDim.x + threadIdx.x) >> 5;
    int lane = threadIdx.x & 31;
    if (pix >= M_TOTAL) return;
    int t = tgt[pix];
    const __bf16* a = A + (size_t)pix * NUM_FEAT + lane * 8;
    const __bf16* l = L + (size_t)t   * NUM_FEAT + lane * 8;
    float s = 0.0f;
#pragma unroll
    for (int i = 0; i < 8; ++i) s += (float)a[i] * (float)l[i];
#pragma unroll
    for (int m = 16; m; m >>= 1) s += __shfl_xor(s, m, 32);
    if (lane == 0) tl[pix] = s;
}

// (4) fused GEMM + online softmax.
// Workgroup = 256 threads = 8 waves; wave owns 16 pixel rows for ALL of N.
// LUT tiles are double-buffered in LDS; tile nt+1 streams in via async
// global->LDS DMA (no VGPR staging) while tile nt's WMMAs run; tile nt+2 is
// prefetched (global_prefetch_b8) toward the WGP.
__global__ __launch_bounds__(256)
void oim_main(const __bf16* __restrict__ A, const __bf16* __restrict__ L,
              const float* __restrict__ tl, float* __restrict__ nll) {
    __shared__ __align__(32) __bf16 lsB[2 * 64 * NUM_FEAT];   // 64 KB, ping-pong

    const int lane = threadIdx.x & 31;
    const int wave = threadIdx.x >> 5;
    const int slab = blockIdx.x * 128 + wave * 16;        // first pixel row of this wave
    const int half = lane >> 4;                           // 0: lanes 0-15, 1: lanes 16-31
    const int l16  = lane & 15;

    // --- A fragments for 16 rows x K=256, register resident (8 frags) ---
    // 16-bit A 16x32 layout (matches B): lane holds row M=l16; K = kbase + e,
    // kbase = half*16  ->  two contiguous 16B chunks per frag per lane.
    v16bf afrag[8];
    {
        const __bf16* ab = A + (size_t)(slab + l16) * NUM_FEAT + half * 16;
#pragma unroll
        for (int kc = 0; kc < 8; ++kc) {
            v8bf lo = *(const v8bf*)(ab + kc * 32);
            v8bf hi = *(const v8bf*)(ab + kc * 32 + 8);
            v16bf f;
#pragma unroll
            for (int e = 0; e < 8; ++e) { f[e] = lo[e]; f[e + 8] = hi[e]; }
            afrag[kc] = f;
        }
    }

    // Online softmax state per C-tile VGPR r (rows slab+r and slab+r+8),
    // uniform within each 16-lane half.
    float run_m[8], run_s[8];
#pragma unroll
    for (int r = 0; r < 8; ++r) { run_m[r] = -1e30f; run_s[r] = 0.0f; }

    // --- prologue: tile 0 -> buf0 ---
    tile_copy(L, lsB, threadIdx.x);
    tile_copy_wait();
    __syncthreads();

    for (int nt = 0; nt < NTILES; ++nt) {
        // --- kick off tile nt+1 into the other buffer; overlaps WMMAs below ---
        if (nt + 1 < NTILES)
            tile_copy(L + (size_t)(nt + 1) * 64 * NUM_FEAT,
                      lsB + (size_t)((nt + 1) & 1) * 64 * NUM_FEAT, threadIdx.x);
        // --- prefetch tile nt+2 (256 threads x 128B stride = 32KB) ---
        if (nt + 2 < NTILES)
            __builtin_prefetch((const char*)(L + (size_t)(nt + 2) * 64 * NUM_FEAT)
                               + threadIdx.x * 128, 0, 1);

        // --- 16x64 logits via 32 WMMAs (4 col subtiles x 8 K-chunks) ---
        const __bf16* lb = lsB + (size_t)(nt & 1) * 64 * NUM_FEAT;
        v8f acc[4] = {};
#pragma unroll
        for (int kc = 0; kc < 8; ++kc) {
#pragma unroll
            for (int t = 0; t < 4; ++t) {
                // 16-bit B 32x16, same k order as A: lane holds col N=l16; K = kc*32 + half*16 + e
                // -> one aligned 32B run per lane.
                const v16bf b = *(const v16bf*)(lb + (t * 16 + l16) * NUM_FEAT
                                                + kc * 32 + half * 16);
                acc[t] = __builtin_amdgcn_wmma_f32_16x16x32_bf16(
                    false, afrag[kc], false, b, (short)0, acc[t], false, false);
            }
        }

        // --- online softmax update over these 64 columns ---
        const int nbase = nt * 64;
#pragma unroll
        for (int r = 0; r < 8; ++r) {
            float tm = -1e30f;
#pragma unroll
            for (int t = 0; t < 4; ++t) {
                bool valid = (nbase + t * 16 + l16) < NUM_CLASSES;
                tm = fmaxf(tm, valid ? acc[t][r] : -1e30f);
            }
#pragma unroll
            for (int m = 1; m < 16; m <<= 1) tm = fmaxf(tm, __shfl_xor(tm, m, 32));
            float mn = fmaxf(run_m[r], tm);
            float ts = 0.0f;
#pragma unroll
            for (int t = 0; t < 4; ++t) {
                bool valid = (nbase + t * 16 + l16) < NUM_CLASSES;
                ts += valid ? __expf(acc[t][r] - mn) : 0.0f;
            }
#pragma unroll
            for (int m = 1; m < 16; m <<= 1) ts += __shfl_xor(ts, m, 32);
            run_s[r] = run_s[r] * __expf(run_m[r] - mn) + ts;
            run_m[r] = mn;
        }

        // --- next tile must be fully in LDS before anyone reads it ---
        tile_copy_wait();
        __syncthreads();   // single barrier/iter (safe with ping-pong buffers)
    }

    // --- epilogue: nll[row] = m + log(s) - target_logit ---
    if (l16 == 0) {
#pragma unroll
        for (int r = 0; r < 8; ++r) {
            int row = slab + half * 8 + r;
            nll[row] = run_m[r] + __logf(run_s[r]) - tl[row];
        }
    }
}

// (5) deterministic single-block tree reduction -> mean
__global__ void oim_reduce(const float* __restrict__ nll, float* __restrict__ out) {
    __shared__ float sm[256];
    float s = 0.0f;
    for (int i = threadIdx.x; i < M_TOTAL; i += 256) s += nll[i];
    sm[threadIdx.x] = s;
    __syncthreads();
    for (int k = 128; k; k >>= 1) {
        if (threadIdx.x < k) sm[threadIdx.x] += sm[threadIdx.x + k];
        __syncthreads();
    }
    if (threadIdx.x == 0) out[0] = sm[0] / (float)M_TOTAL;
}

extern "C" void kernel_launch(void* const* d_in, const int* in_sizes, int n_in,
                              void* d_out, int out_size, void* d_ws, size_t ws_size,
                              hipStream_t stream) {
    const float* lut     = (const float*)d_in[0];   // [5532,256]
    const float* inputs  = (const float*)d_in[1];   // [4,256,64,64]
    const int*   targets = (const int*)d_in[2];     // [4,64,64]
    // d_in[3] = epoch (unused)
    float* out = (float*)d_out;

    char* ws = (char*)d_ws;
    __bf16* A  = (__bf16*)ws;                                              // 8 MB
    __bf16* L  = (__bf16*)(ws + (size_t)M_TOTAL * NUM_FEAT * 2);           // 2.78 MB
    float*  tl = (float*)(ws + (size_t)M_TOTAL * NUM_FEAT * 2
                             + (size_t)NPAD * NUM_FEAT * 2);               // 64 KB
    float* nll = tl + M_TOTAL;                                             // 64 KB

    oim_prep_a  <<<(4 * NUM_FEAT * HW) / 256, 256, 0, stream>>>(inputs, A);
    oim_prep_lut<<<(NPAD * NUM_FEAT) / 256,   256, 0, stream>>>(lut, L);
    oim_tlogit  <<<(M_TOTAL * 32) / 256,      256, 0, stream>>>(A, L, targets, tl);
    oim_main    <<<M_TOTAL / 128,             256, 0, stream>>>(A, L, tl, nll);
    oim_reduce  <<<1,                         256, 0, stream>>>(nll, out);
}
